// HierarchicalHAMTModel_13271448944698
// MI455X (gfx1250) — matrix operand order repaired and checked by reading, not compiled, not run on hardware
//
#include <hip/hip_runtime.h>
#include <math.h>

// ---------------- types ----------------
typedef _Float16 f16;
typedef _Float16 v16h __attribute__((ext_vector_type(16)));
typedef _Float16 v8h  __attribute__((ext_vector_type(8)));
typedef float    v8f  __attribute__((ext_vector_type(8)));
typedef float    v4f  __attribute__((ext_vector_type(4)));

union FragA { v16h v; v8h h[2]; };

// model dims
static constexpr int cS  = 2048;
static constexpr int cV  = 32000;
static constexpr int cH  = 1024;
static constexpr int cD  = 512;
static constexpr int cI  = 4096;
static constexpr int cL  = 2;
static constexpr int cSL = 32;

enum { ACT_NONE = 0, ACT_GELU = 1, ACT_SIGMOID = 2 };

// ---------------- WMMA GEMM ----------------
// C[M,N] = act( A[M,K] @ B + bias )   (fp32 in/out, f16 WMMA, f32 accum)
// BT=false: B is KxN row-major.   BT=true: B is NxK row-major (C = A @ B^T).
template <bool BT>
__global__ __launch_bounds__(256) void gemm_wmma_kernel(
    const float* __restrict__ A, const float* __restrict__ B,
    const float* __restrict__ bias, float* __restrict__ C,
    int M, int N, int K, int act)
{
    constexpr int BM = 128, BN = 128, BK = 32;
    constexpr int LDA = 40, LDB = 40;   // f16 units, padded vs 32 -> conflict-free frag reads
    __shared__ f16 sA[BM * LDA];
    __shared__ f16 sB[BN * LDB];

    const int m0   = blockIdx.y * BM;
    const int n0   = blockIdx.x * BN;
    const int tid  = threadIdx.x;
    const int lane = tid & 31;
    const int w    = tid >> 5;          // 0..7
    const int wm   = (w >> 2) * 64;     // wave row offset: 0 / 64
    const int wn   = (w & 3) * 32;      // wave col offset: 0..96
    const int kh   = lane >> 4;         // lane half (0/1)
    const int lm   = lane & 15;

    v8f acc[4][2] = {};

    for (int k0 = 0; k0 < K; k0 += BK) {
        // ---- stage A tile (BMxBK), f32 -> f16 ----
        {
            const int r = tid >> 3;            // 0..31
            const int c = (tid & 7) * 4;       // 0..28
            #pragma unroll
            for (int rr = r; rr < BM; rr += 32) {
                const int gm = m0 + rr;
                v4f v = {};
                if (gm < M) {
                    v = *(const v4f*)(A + (size_t)gm * K + k0 + c);
                    if (k0 + BK < K) __builtin_prefetch(A + (size_t)gm * K + k0 + BK + c, 0, 1);
                }
                f16* d = &sA[rr * LDA + c];
                d[0] = (f16)v.x; d[1] = (f16)v.y; d[2] = (f16)v.z; d[3] = (f16)v.w;
            }
        }
        // ---- stage B tile into sB[n][k] ----
        if (BT) {
            const int r = tid >> 3;
            const int c = (tid & 7) * 4;
            #pragma unroll
            for (int rr = r; rr < BN; rr += 32) {
                const int gn = n0 + rr;
                v4f v = {};
                if (gn < N) v = *(const v4f*)(B + (size_t)gn * K + k0 + c);
                f16* d = &sB[rr * LDB + c];
                d[0] = (f16)v.x; d[1] = (f16)v.y; d[2] = (f16)v.z; d[3] = (f16)v.w;
            }
        } else {
            const int kk = tid >> 6;           // 0..3
            const int nn = (tid & 63) * 2;     // 0..126
            #pragma unroll
            for (int k2 = kk; k2 < BK; k2 += 4) {
                float x0 = 0.f, x1 = 0.f;
                if (n0 + nn + 1 < N) {
                    const float2 v = *(const float2*)(B + (size_t)(k0 + k2) * N + n0 + nn);
                    x0 = v.x; x1 = v.y;
                } else if (n0 + nn < N) {
                    x0 = B[(size_t)(k0 + k2) * N + n0 + nn];
                }
                sB[(nn    ) * LDB + k2] = (f16)x0;
                sB[(nn + 1) * LDB + k2] = (f16)x1;
            }
        }
        __syncthreads();

        // ---- fragments per ISA 7.12.2 f16 layouts ----
        FragA af[4], bf[2];
        #pragma unroll
        for (int mi = 0; mi < 4; mi++) {
            const f16* p = &sA[(wm + mi * 16 + lm) * LDA + 8 * kh];
            af[mi].h[0] = *(const v8h*)p;           // K = 8*kh .. +7
            af[mi].h[1] = *(const v8h*)(p + 16);    // K = 16+8*kh .. +7
        }
        #pragma unroll
        for (int ni = 0; ni < 2; ni++) {
            const f16* p = &sB[(wn + ni * 16 + lm) * LDB + 8 * kh];
            bf[ni].h[0] = *(const v8h*)p;           // K = 8*kh .. +7
            bf[ni].h[1] = *(const v8h*)(p + 16);    // K = 16+8*kh .. +7
        }
        #pragma unroll
        for (int mi = 0; mi < 4; mi++)
            #pragma unroll
            for (int ni = 0; ni < 2; ni++)
                acc[mi][ni] = __builtin_amdgcn_wmma_f32_16x16x32_f16(
                    false, af[mi].v, false, bf[ni].v, (short)0, acc[mi][ni], false, false);
        __syncthreads();
    }

    // ---- epilogue: bias + activation + store ----
    #pragma unroll
    for (int mi = 0; mi < 4; mi++) {
        #pragma unroll
        for (int ni = 0; ni < 2; ni++) {
            #pragma unroll
            for (int r = 0; r < 8; r++) {
                const int gm = m0 + wm + mi * 16 + r + 8 * kh;
                const int gn = n0 + wn + ni * 16 + lm;
                if (gm < M && gn < N) {
                    float x = acc[mi][ni][r];
                    if (bias) x += bias[gn];
                    if (act == ACT_GELU) {
                        const float t = 0.7978845608028654f * (x + 0.044715f * x * x * x);
                        x = 0.5f * x * (1.0f + tanhf(t));
                    } else if (act == ACT_SIGMOID) {
                        x = 1.0f / (1.0f + expf(-x));
                    }
                    C[(size_t)gm * N + gn] = x;
                }
            }
        }
    }
}

// ---------------- small kernels ----------------
__global__ void embed_kernel(const int* __restrict__ ids, const float* __restrict__ emb,
                             const float* __restrict__ pos, float* __restrict__ h)
{
    const int s = blockIdx.x;
    const int v = ids[s];
    for (int i = threadIdx.x; i < cH; i += blockDim.x)
        h[(size_t)s * cH + i] = emb[(size_t)v * cH + i] + pos[(size_t)s * cH + i];
}

__device__ __forceinline__ float block_reduce_sum256(float v, float* red)
{
    #pragma unroll
    for (int o = 16; o > 0; o >>= 1) v += __shfl_down(v, o, 32);
    const int wv = threadIdx.x >> 5;
    if ((threadIdx.x & 31) == 0) red[wv] = v;
    __syncthreads();
    float t = (threadIdx.x < 8) ? red[threadIdx.x] : 0.0f;
    #pragma unroll
    for (int o = 4; o > 0; o >>= 1) t += __shfl_down(t, o, 32);
    if (threadIdx.x == 0) red[0] = t;
    __syncthreads();
    const float r = red[0];
    __syncthreads();
    return r;
}

// keys[s,:] /= (||keys[s,:]|| + 1e-6), row length cD
__global__ void norm_rows_kernel(float* __restrict__ x)
{
    __shared__ float red[8];
    float* row = x + (size_t)blockIdx.x * cD;
    float ss = 0.f;
    for (int i = threadIdx.x; i < cD; i += 256) { const float v = row[i]; ss += v * v; }
    const float tot = block_reduce_sum256(ss, red);
    const float inv = 1.0f / (sqrtf(tot) + 1e-6f);
    for (int i = threadIdx.x; i < cD; i += 256) row[i] *= inv;
}

// scores over 64 memory slots -> softmax -> retrieved = (attn @ mem) * keys
__global__ void attn_kernel(const float* __restrict__ keys, const float* __restrict__ fast0,
                            const float* __restrict__ slow0, float* __restrict__ retr)
{
    __shared__ float krow[cD];
    __shared__ float attn[64];
    const int s = blockIdx.x;
    for (int i = threadIdx.x; i < cD; i += 256) krow[i] = keys[(size_t)s * cD + i];
    __syncthreads();
    if (threadIdx.x < 64) {
        const int n = threadIdx.x;
        const float* mrow = (n < cSL) ? (fast0 + (size_t)n * cD) : (slow0 + (size_t)(n - cSL) * cD);
        float d = 0.f;
        for (int i = 0; i < cD; i++) d += krow[i] * mrow[i];
        attn[n] = d * 0.04419417382415922f;   // 1/sqrt(512)
    }
    __syncthreads();
    if (threadIdx.x == 0) {
        float mx = attn[0];
        for (int i = 1; i < 64; i++) mx = fmaxf(mx, attn[i]);
        float sm = 0.f;
        for (int i = 0; i < 64; i++) { const float e = expf(attn[i] - mx); attn[i] = e; sm += e; }
        const float inv = 1.0f / sm;
        for (int i = 0; i < 64; i++) attn[i] *= inv;
    }
    __syncthreads();
    for (int d = threadIdx.x; d < cD; d += 256) {
        float acc = 0.f;
        #pragma unroll 4
        for (int n = 0; n < cSL; n++) acc += attn[n]        * fast0[(size_t)n * cD + d];
        #pragma unroll 4
        for (int n = 0; n < cSL; n++) acc += attn[cSL + n]  * slow0[(size_t)n * cD + d];
        retr[(size_t)s * cD + d] = acc * krow[d];
    }
}

__global__ void concat_kernel(const float* __restrict__ a, const float* __restrict__ b,
                              float* __restrict__ out, int na, int nb)
{
    const int s = blockIdx.x;
    const int nt = na + nb;
    for (int i = threadIdx.x; i < nt; i += blockDim.x)
        out[(size_t)s * nt + i] = (i < na) ? a[(size_t)s * na + i] : b[(size_t)s * nb + (i - na)];
}

// sequential memory scan: one block, thread d owns column d of fast/slow (32 rows each)
__global__ __launch_bounds__(512) void scan_kernel(
    const float* __restrict__ items, const float* __restrict__ gates,
    const float* __restrict__ fast0, const float* __restrict__ slow0,
    float* __restrict__ fastOut, float* __restrict__ slowOut)
{
    __shared__ float g[cSL];
    const int d = threadIdx.x;   // 0..511
    float fast[cSL], slow[cSL];
    #pragma unroll
    for (int n = 0; n < cSL; n++) {
        fast[n] = fast0[(size_t)n * cD + d];
        slow[n] = slow0[(size_t)n * cD + d];
    }
    for (int s = 0; s < cS; s++) {
        __syncthreads();
        if (d < cSL) g[d] = gates[(size_t)s * (2 * cSL) + d];   // fast_gates = gates[..., :SL]
        __syncthreads();
        const float it = items[(size_t)s * cD + d];
        const bool c = (s % 10) == 0;
        #pragma unroll
        for (int n = 0; n < cSL; n++) {
            float f = 0.9f * fast[n] + g[n] * it;
            if (c) { slow[n] = 0.99f * slow[n] + 0.1f * f; f *= 0.9f; }
            fast[n] = f;
        }
    }
    #pragma unroll
    for (int n = 0; n < cSL; n++) {
        fastOut[(size_t)n * cD + d] = fast[n];
        slowOut[(size_t)n * cD + d] = slow[n];
    }
}

// x[s,:] = LN(x[s,:] + res[s,:]) * g + b   (row length cH; res may be null)
__global__ void ln_kernel(float* __restrict__ x, const float* __restrict__ res,
                          const float* __restrict__ g, const float* __restrict__ b)
{
    __shared__ float red[8];
    const int s = blockIdx.x;
    float* xr = x + (size_t)s * cH;
    const float* rr = res ? res + (size_t)s * cH : nullptr;
    float v[4];
    float sum = 0.f;
    #pragma unroll
    for (int j = 0; j < 4; j++) {
        const int i = threadIdx.x + j * 256;
        const float t = xr[i] + (rr ? rr[i] : 0.f);
        v[j] = t; sum += t;
    }
    const float mean = block_reduce_sum256(sum, red) * (1.0f / cH);
    float var = 0.f;
    #pragma unroll
    for (int j = 0; j < 4; j++) { const float dd = v[j] - mean; var += dd * dd; }
    var = block_reduce_sum256(var, red) * (1.0f / cH);
    const float inv = rsqrtf(var + 1e-5f);
    #pragma unroll
    for (int j = 0; j < 4; j++) {
        const int i = threadIdx.x + j * 256;
        xr[i] = (v[j] - mean) * inv * g[i] + b[i];
    }
}

// ---------------- launch ----------------
static inline void gemm_nn(const float* A, const float* B, const float* bias, float* C,
                           int M, int N, int K, int act, hipStream_t st)
{
    dim3 grid((N + 127) / 128, (M + 127) / 128);
    gemm_wmma_kernel<false><<<grid, 256, 0, st>>>(A, B, bias, C, M, N, K, act);
}
static inline void gemm_nt(const float* A, const float* B, const float* bias, float* C,
                           int M, int N, int K, int act, hipStream_t st)
{
    dim3 grid((N + 127) / 128, (M + 127) / 128);
    gemm_wmma_kernel<true><<<grid, 256, 0, st>>>(A, B, bias, C, M, N, K, act);
}

extern "C" void kernel_launch(void* const* d_in, const int* in_sizes, int n_in,
                              void* d_out, int out_size, void* d_ws, size_t ws_size,
                              hipStream_t stream)
{
    (void)in_sizes; (void)n_in; (void)out_size; (void)ws_size;

    const int*   ids     = (const int*)  d_in[0];
    const float* fast_in = (const float*)d_in[1];
    const float* slow_in = (const float*)d_in[2];
    const float* emb     = (const float*)d_in[3];
    const float* pos     = (const float*)d_in[4];
    const float* W_item  = (const float*)d_in[5];
    const float* b_item  = (const float*)d_in[6];
    const float* W_q     = (const float*)d_in[7];
    const float* b_q     = (const float*)d_in[8];
    const float* rh_W1   = (const float*)d_in[9];
    const float* rh_b1   = (const float*)d_in[10];
    const float* rh_W2   = (const float*)d_in[11];
    const float* rh_b2   = (const float*)d_in[12];
    const float* gate_W  = (const float*)d_in[13];
    const float* gate_b  = (const float*)d_in[14];
    const float* W_out   = (const float*)d_in[15];
    const float* b_out   = (const float*)d_in[16];
    const float* ln1_g   = (const float*)d_in[17];
    const float* ln1_b   = (const float*)d_in[18];
    const float* ffn_W1  = (const float*)d_in[19];
    const float* ffn_b1  = (const float*)d_in[20];
    const float* ffn_W2  = (const float*)d_in[21];
    const float* ffn_b2  = (const float*)d_in[22];
    const float* ln2_g   = (const float*)d_in[23];
    const float* ln2_b   = (const float*)d_in[24];
    const float* fln_g   = (const float*)d_in[25];
    const float* fln_b   = (const float*)d_in[26];

    // workspace layout (floats); ffn1 aliases buffers dead by FFN time
    float* ws    = (float*)d_ws;
    float* h     = ws;
    float* items = h     + (size_t)cS * cH;
    float* query = items + (size_t)cS * cD;
    float* k1    = query + (size_t)cS * cH;
    float* keys  = k1    + (size_t)cS * cH;
    float* retr  = keys  + (size_t)cS * cD;
    float* hr    = retr  + (size_t)cS * cD;
    float* gates = hr    + (size_t)cS * (cH + cD);
    float* outb  = gates + (size_t)cS * (2 * cSL);
    float* ffn1  = items;   // alias: items/query/k1/keys/retr/hr all dead during FFN

    float* logits = (float*)d_out;
    float* ofast  = logits + (size_t)cS * cV;
    float* oslow  = ofast  + (size_t)cL * cSL * cD;

    embed_kernel<<<cS, 256, 0, stream>>>(ids, emb, pos, h);

    for (int l = 0; l < cL; l++) {
        const float* f0 = fast_in + (size_t)l * cSL * cD;
        const float* s0 = slow_in + (size_t)l * cSL * cD;

        gemm_nn(h, W_item + (size_t)l * cH * cD, b_item + (size_t)l * cD,
                items, cS, cD, cH, ACT_NONE, stream);
        gemm_nn(h, W_q + (size_t)l * cH * cH, b_q + (size_t)l * cH,
                query, cS, cH, cH, ACT_NONE, stream);
        gemm_nn(items, rh_W1 + (size_t)l * cD * 2 * cD, rh_b1 + (size_t)l * 2 * cD,
                k1, cS, 2 * cD, cD, ACT_GELU, stream);
        gemm_nn(k1, rh_W2 + (size_t)l * 2 * cD * cD, rh_b2 + (size_t)l * cD,
                keys, cS, cD, 2 * cD, ACT_NONE, stream);
        norm_rows_kernel<<<cS, 256, 0, stream>>>(keys);
        attn_kernel<<<cS, 256, 0, stream>>>(keys, f0, s0, retr);

        concat_kernel<<<cS, 256, 0, stream>>>(h, retr, hr, cH, cD);
        gemm_nn(hr, gate_W + (size_t)l * (cH + cD) * (2 * cSL), gate_b + (size_t)l * 2 * cSL,
                gates, cS, 2 * cSL, cH + cD, ACT_SIGMOID, stream);

        scan_kernel<<<1, 512, 0, stream>>>(items, gates, f0, s0,
                                           ofast + (size_t)l * cSL * cD,
                                           oslow + (size_t)l * cSL * cD);

        concat_kernel<<<cS, 256, 0, stream>>>(query, retr, hr, cH, cD);
        gemm_nn(hr, W_out + (size_t)l * (cH + cD) * cH, b_out + (size_t)l * cH,
                outb, cS, cH, cH + cD, ACT_NONE, stream);
        ln_kernel<<<cS, 256, 0, stream>>>(h, outb, ln1_g + (size_t)l * cH, ln1_b + (size_t)l * cH);

        gemm_nn(h, ffn_W1 + (size_t)l * cH * cI, ffn_b1 + (size_t)l * cI,
                ffn1, cS, cI, cH, ACT_GELU, stream);
        gemm_nn(ffn1, ffn_W2 + (size_t)l * cI * cH, ffn_b2 + (size_t)l * cH,
                outb, cS, cH, cI, ACT_NONE, stream);
        ln_kernel<<<cS, 256, 0, stream>>>(h, outb, ln2_g + (size_t)l * cH, ln2_b + (size_t)l * cH);
    }

    ln_kernel<<<cS, 256, 0, stream>>>(h, nullptr, fln_g, fln_b);
    gemm_nt(h, emb, nullptr, logits, cS, cV, cH, ACT_NONE, stream);
}
